// TextSemanticEnrichment_42039139893959
// MI455X (gfx1250) — matrix operand order repaired and checked by reading, not compiled, not run on hardware
//
#include <hip/hip_runtime.h>

// ---------------------------------------------------------------------------
// Problem constants (from reference): B=131072 rows, D=512, K=112 centers, top-5
// Roofline: 512 MB mandatory HBM traffic -> ~22us @ 23.3 TB/s. GEMM needs
// ~680 TFLOPS to hide under that -> must use v_wmma_f32_16x16x32_bf16.
// Ranking in bf16, scoring/aggregation exact fp32 (reads are L2-hot).
// ---------------------------------------------------------------------------

#define B_ROWS   131072
#define D_DIM    512
#define K_CLUST  112
#define TOPK     5
#define LAMBDA_  5.0f

#define NT       7     // 112 / 16 column tiles
#define KSTEPS   16    // 512 / 32 k-steps
#define ROWS_PB  128   // rows per block (8 waves x 16 rows)
#define THREADS  256

typedef __attribute__((ext_vector_type(16))) __bf16 v16bf;
typedef __attribute__((ext_vector_type(8)))  float  v8f;

// ---------------------------------------------------------------------------
// Pack centers fp32 -> bf16 into d_ws in the per-lane WMMA B-operand
// layout used here (contiguous K, same as A; 16-bit B 32x16: lanes 0-15 hold K=0..15, lanes 16-31
// hold K=16..31, N = lane&15, K packed 2-per-VGPR in ascending order).
// Fragment for (tile t, kstep kb) occupies 32 lanes x 32 bytes, so the main
// kernel fetches one v16bf per lane with a single coalesced 32B load.
// Needs NT*KSTEPS*32*16*2 = 114,688 bytes of d_ws.
// ---------------------------------------------------------------------------
__global__ void pack_centers_bf16(const float* __restrict__ centers,
                                  __bf16* __restrict__ wsB) {
    int id = blockIdx.x * blockDim.x + threadIdx.x;   // (t, kb, lane)
    if (id >= NT * KSTEPS * 32) return;
    int lane = id & 31;
    int kb   = (id >> 5) & (KSTEPS - 1);
    int t    = id / (KSTEPS * 32);
    int n     = t * 16 + (lane & 15);                 // cluster index (N)
    int kbase = kb * 32 + ((lane >> 4) << 4);         // 16 contiguous K values
    const float* src = centers + (size_t)n * D_DIM + kbase;
    __bf16* dst = wsB + (size_t)id * 16;
#pragma unroll
    for (int e = 0; e < 16; ++e) dst[e] = (__bf16)src[e];
}

// ---------------------------------------------------------------------------
// Fused: WMMA sims tile -> LDS -> top-5 -> exact fp32 rescore -> softmax ->
// softmax-weighted aggregation + residual.
// ---------------------------------------------------------------------------
__global__ __launch_bounds__(THREADS)
void enrich_kernel(const float* __restrict__ x,
                   const float* __restrict__ centers,
                   const __bf16* __restrict__ wsB,
                   float* __restrict__ out) {
    __shared__ float sims[ROWS_PB][K_CLUST + 1];   // +1 pad vs bank conflicts
    __shared__ int   sIdx[ROWS_PB][TOPK];
    __shared__ float sScr[ROWS_PB][TOPK];

    const int tid      = threadIdx.x;
    const int lane     = tid & 31;
    const int wave     = tid >> 5;
    const int blockRow = blockIdx.x * ROWS_PB;

    // ---------------- Phase 1: sims[16x112] per wave via bf16 WMMA ---------
    v8f acc[NT];
    const v8f vzero = {0.f, 0.f, 0.f, 0.f, 0.f, 0.f, 0.f, 0.f};
#pragma unroll
    for (int t = 0; t < NT; ++t) acc[t] = vzero;

    // A fragment (16x32 bf16): lane half selects K 0-15 / 16-31,
    // vector elements 0..15 are 16 contiguous K values.
    const int aRow  = blockRow + wave * 16 + (lane & 15);
    const int kHalf = (lane >> 4) << 4;               // 0 or 16
    const float* xrow = x + (size_t)aRow * D_DIM;

    for (int kb = 0; kb < KSTEPS; ++kb) {
        const float* xp = xrow + kb * 32 + kHalf;
        float4 f0 = *(const float4*)(xp + 0);
        float4 f1 = *(const float4*)(xp + 4);
        float4 f2 = *(const float4*)(xp + 8);
        float4 f3 = *(const float4*)(xp + 12);
        v16bf a;
        a[0]  = (__bf16)f0.x; a[1]  = (__bf16)f0.y; a[2]  = (__bf16)f0.z; a[3]  = (__bf16)f0.w;
        a[4]  = (__bf16)f1.x; a[5]  = (__bf16)f1.y; a[6]  = (__bf16)f1.z; a[7]  = (__bf16)f1.w;
        a[8]  = (__bf16)f2.x; a[9]  = (__bf16)f2.y; a[10] = (__bf16)f2.z; a[11] = (__bf16)f2.w;
        a[12] = (__bf16)f3.x; a[13] = (__bf16)f3.y; a[14] = (__bf16)f3.z; a[15] = (__bf16)f3.w;
#pragma unroll
        for (int t = 0; t < NT; ++t) {
            v16bf b = *(const v16bf*)(wsB + ((size_t)((t * KSTEPS + kb) * 32 + lane)) * 16);
            // 8 args: (neg_a, A, neg_b, B, c_mod, C, reuse_a, reuse_b)
            acc[t] = __builtin_amdgcn_wmma_f32_16x16x32_bf16(
                false, a, false, b, (short)0, acc[t], false, false);
        }
    }

    // C layout: VGPR j, lane l -> row = j + (l>=16 ? 8 : 0), col = l&15.
    const int rBase = wave * 16 + ((lane >> 4) << 3);
    const int col   = lane & 15;
#pragma unroll
    for (int t = 0; t < NT; ++t) {
#pragma unroll
        for (int j = 0; j < 8; ++j)
            sims[rBase + j][t * 16 + col] = acc[t][j];
    }
    __syncthreads();

    // ---------------- Phase 2: top-5 + exact fp32 rescore + softmax --------
    if (tid < ROWS_PB) {
        const int r = tid;
        float bestV[TOPK];
        int   bestI[TOPK];
#pragma unroll
        for (int k = 0; k < TOPK; ++k) { bestV[k] = -3.0e38f; bestI[k] = 0; }
        for (int c = 0; c < K_CLUST; ++c) {
            float v = sims[r][c];
            if (v > bestV[TOPK - 1]) {            // strict >: jax tie order
                int p = TOPK - 1;
                while (p > 0 && v > bestV[p - 1]) {
                    bestV[p] = bestV[p - 1]; bestI[p] = bestI[p - 1]; --p;
                }
                bestV[p] = v; bestI[p] = c;
            }
        }
        // Exact fp32 logits (reference recomputes einsum in fp32); x row and
        // centers are L2-hot so this costs no HBM traffic.
        const float4* xr4 = (const float4*)(x + (size_t)(blockRow + r) * D_DIM);
        const float4* cp[TOPK];
#pragma unroll
        for (int k = 0; k < TOPK; ++k)
            cp[k] = (const float4*)(centers + (size_t)bestI[k] * D_DIM);
        float dot[TOPK] = {0.f, 0.f, 0.f, 0.f, 0.f};
        for (int q = 0; q < D_DIM / 4; ++q) {
            float4 xv = xr4[q];
#pragma unroll
            for (int k = 0; k < TOPK; ++k) {
                float4 cv = cp[k][q];
                dot[k] += xv.x * cv.x + xv.y * cv.y + xv.z * cv.z + xv.w * cv.w;
            }
        }
        float m = dot[0];
#pragma unroll
        for (int k = 1; k < TOPK; ++k) m = fmaxf(m, dot[k]);
        float e[TOPK], sum = 0.f;
#pragma unroll
        for (int k = 0; k < TOPK; ++k) { e[k] = __expf(dot[k] - m); sum += e[k]; }
        float inv = 1.0f / sum;
#pragma unroll
        for (int k = 0; k < TOPK; ++k) { sIdx[r][k] = bestI[k]; sScr[r][k] = e[k] * inv; }
    }
    __syncthreads();

    // ---------------- Phase 3: out = x + 5 * sum_k s_k * C[idx_k] ----------
    {
        const int r    = tid >> 1;        // 2 threads per row
        const int half = tid & 1;         // each covers 64 float4 = 256 floats
        const int row  = blockRow + r;
        float s[TOPK];
        const float4* c4[TOPK];
#pragma unroll
        for (int k = 0; k < TOPK; ++k) {
            s[k]  = sScr[r][k];
            c4[k] = (const float4*)(centers + (size_t)sIdx[r][k] * D_DIM) + half * 64;
        }
        const float4* xr4 = (const float4*)(x + (size_t)row * D_DIM) + half * 64;
        float4*       o4  = (float4*)(out + (size_t)row * D_DIM) + half * 64;
        for (int q = 0; q < 64; ++q) {
            float4 xv = xr4[q];
            float4 ag = {0.f, 0.f, 0.f, 0.f};
#pragma unroll
            for (int k = 0; k < TOPK; ++k) {
                float4 cv = c4[k][q];
                ag.x += s[k] * cv.x; ag.y += s[k] * cv.y;
                ag.z += s[k] * cv.z; ag.w += s[k] * cv.w;
            }
            float4 res;
            res.x = xv.x + LAMBDA_ * ag.x;
            res.y = xv.y + LAMBDA_ * ag.y;
            res.z = xv.z + LAMBDA_ * ag.z;
            res.w = xv.w + LAMBDA_ * ag.w;
            o4[q] = res;
        }
    }
}

// ---------------------------------------------------------------------------
extern "C" void kernel_launch(void* const* d_in, const int* in_sizes, int n_in,
                              void* d_out, int out_size, void* d_ws, size_t ws_size,
                              hipStream_t stream) {
    const float* x       = (const float*)d_in[0];   // [131072, 512] fp32
    const float* centers = (const float*)d_in[1];   // [112, 512] fp32
    // d_in[2] = topk scalar (5), fixed at compile time.
    __bf16* wsB = (__bf16*)d_ws;                    // needs 114,688 bytes
    float*  out = (float*)d_out;

    pack_centers_bf16<<<(NT * KSTEPS * 32 + 255) / 256, 256, 0, stream>>>(centers, wsB);
    enrich_kernel<<<B_ROWS / ROWS_PB, THREADS, 0, stream>>>(x, centers, wsB, out);
}
